// ResnetBlocWithAttn_2791728742776
// MI455X (gfx1250) — matrix operand order repaired and checked by reading, not compiled, not run on hardware
//
#include <hip/hip_runtime.h>

// ---------------------------------------------------------------------------
// ResnetBlockWithAttn for MI455X (gfx1250): bf16 WMMA everywhere, fp32 stats.
// B=16, C=256, H=W=32, E=256, groups=32, 1 attention head.
// All GEMM-fed activations stored token-major [b][hw][C] (bf16 as ushort).
// Attention K/V chunks staged into LDS by the Tensor Data Mover (double-
// buffered), WMMAs fed from LDS instead of latency-bound global loads.
// ---------------------------------------------------------------------------

typedef unsigned short ushort_t;
typedef __attribute__((ext_vector_type(16))) __bf16 bf16x16;
typedef __attribute__((ext_vector_type(8)))  float  v8f;
typedef unsigned int __attribute__((ext_vector_type(4))) uint32x4_t;
typedef int          __attribute__((ext_vector_type(8))) int32x8_t;
typedef int          __attribute__((ext_vector_type(4))) int32x4_t;

#if defined(__has_builtin)
#  if __has_builtin(__builtin_amdgcn_tensor_load_to_lds)
#    define HAVE_TDM 1
#  endif
#endif
#ifndef HAVE_TDM
#  define HAVE_TDM 0
#endif

union AFrag { bf16x16 v; uint4 u[2]; };

static __device__ __forceinline__ ushort_t f32_to_bf16u(float f) {
  unsigned int u = __float_as_uint(f);
  u += 0x7FFFu + ((u >> 16) & 1u);          // round-to-nearest-even
  return (ushort_t)(u >> 16);
}

static __device__ __forceinline__ bf16x16 zero16() {
  AFrag f; f.u[0] = make_uint4(0,0,0,0); f.u[1] = make_uint4(0,0,0,0); return f.v;
}
static __device__ __forceinline__ v8f zero8() {
  v8f z = {0.f,0.f,0.f,0.f,0.f,0.f,0.f,0.f}; return z;
}

// A fragment (16x32 bf16, M=row striped in-lane):
// lanes 0-15: K = kbase+{0..15}; lanes 16-31: K = kbase+{16..31}
static __device__ __forceinline__ bf16x16 load_a16(const ushort_t* row, int kbase, int lane) {
  int ko = kbase + ((lane & 16) ? 16 : 0);
  AFrag f;
  f.u[0] = *(const uint4*)(row + ko);
  f.u[1] = *(const uint4*)(row + ko + 8);
  return f.v;
}
// B fragment (32x16 bf16): lane holds column N=lane&15, 16 contiguous K values
// starting at ksbase (+16 for lanes 16-31). rowN points at that column's
// K-contiguous row in memory.
static __device__ __forceinline__ bf16x16 load_b16(const ushort_t* rowN, int ksbase, int lane) {
  int ks = ksbase + ((lane & 16) ? 16 : 0);
  AFrag f;
  f.u[0] = *(const uint4*)(rowN + ks);
  f.u[1] = *(const uint4*)(rowN + ks + 8);
  return f.v;
}
// Pre-packed weight fragment: 16 contiguous bf16 per lane.
static __device__ __forceinline__ bf16x16 load_wpack(const ushort_t* p) {
  AFrag f; f.u[0] = *(const uint4*)p; f.u[1] = *(const uint4*)(p + 8); return f.v;
}
static __device__ __forceinline__ v8f wmma_bf16(bf16x16 a, bf16x16 b, v8f c) {
  return __builtin_amdgcn_wmma_f32_16x16x32_bf16(false, a, false, b, (short)0, c, false, false);
}

#if HAVE_TDM
// Issue a 2-D TDM tile load (global -> LDS). Sizes/strides in 8-byte units.
// D# group0: count=1 | lds_addr | global_addr[56:0] | type=2 ("image").
// D# group1: data_size=8B; tensor dims == tile dims (no OOB); dim0 row stride.
// This toolchain's builtin takes 6 args: (g0, g1, g2, g3, g4, cpol).
static __device__ __forceinline__ void tdm_load_2d(unsigned lds_off, const void* gptr,
                                                   unsigned tile_d0_u8, unsigned tile_d1,
                                                   unsigned row_stride_u8) {
  unsigned long long ga = (unsigned long long)(size_t)gptr;
  uint32x4_t g0;
  g0[0] = 1u;                                                  // count=1, user D#
  g0[1] = lds_off;                                             // lds_addr bytes
  g0[2] = (unsigned)ga;                                        // global_addr[31:0]
  g0[3] = (unsigned)((ga >> 32) & 0x01FFFFFFu) | (2u << 30);   // addr[56:32] | type=2
  int32x8_t g1;
  g1[0] = 3 << 16;                     // workgroup_mask=0, data_size=8B, no pad/iter
  g1[1] = (int)(tile_d0_u8 << 16);     // tensor_dim0[15:0]
  g1[2] = (int)(tile_d1 << 16);        // tensor_dim0 hi=0 | tensor_dim1[15:0]
  g1[3] = (int)(tile_d0_u8 << 16);     // tensor_dim1 hi=0 | tile_dim0
  g1[4] = (int)(tile_d1 & 0xFFFFu);    // tile_dim1 | tile_dim2=0 (2-D)
  g1[5] = (int)row_stride_u8;          // tensor_dim0_stride[31:0]
  g1[6] = 0;
  g1[7] = 0;
  int32x4_t z4 = {0, 0, 0, 0};
  int32x8_t z8 = {0, 0, 0, 0, 0, 0, 0, 0};
  __builtin_amdgcn_tensor_load_to_lds(g0, g1, z4, z4, z8, 0);
}
#endif

// ---------------------------------------------------------------------------
// Weight packing: wp[(((tap*8 + kt)*NT + nt)*32 + lane)*16 + j]
//   n = nt*16 + (lane&15);  k = kt*32 + ((lane&16)?16:0) + j
//   w is (O, I=256, kh, kw) fp32 row-major, tap = kh*3 + kw.
// ---------------------------------------------------------------------------
__global__ void pack_w_kernel(const float* __restrict__ w, ushort_t* __restrict__ wp,
                              int NT, int TAPS) {
  int id = blockIdx.x * blockDim.x + threadIdx.x;
  int total = TAPS * 8 * NT * 32 * 16;
  if (id >= total) return;
  int j    = id & 15;
  int lane = (id >> 4) & 31;
  int rest = id >> 9;
  int nt   = rest % NT;  rest /= NT;
  int kt   = rest & 7;
  int tap  = rest >> 3;
  int n = nt * 16 + (lane & 15);
  int k = kt * 32 + ((lane & 16) ? 16 : 0) + j;
  wp[id] = f32_to_bf16u(w[(n * 256 + k) * TAPS + tap]);
}

// ---------------------------------------------------------------------------
// temb[b][c] = time_emb[b] . noise_w[c] + noise_b[c]     (16 x 256 x 256)
// ---------------------------------------------------------------------------
__global__ void time_proj_kernel(const float* __restrict__ emb, const float* __restrict__ w,
                                 const float* __restrict__ bias, float* __restrict__ temb) {
  int b = blockIdx.x, c = threadIdx.x;
  const float* e  = emb + b * 256;
  const float* wr = w + c * 256;
  float s = 0.f;
  #pragma unroll 8
  for (int i = 0; i < 256; ++i) s += e[i] * wr[i];
  temb[b * 256 + c] = s + bias[c];
}

// ---------------------------------------------------------------------------
// GroupNorm stats: one block per (b, g); 8 channels x 1024 spatial = 8192 elems
// ---------------------------------------------------------------------------
__global__ void gn_stats_kernel(const float* __restrict__ in, float* __restrict__ mean,
                                float* __restrict__ rstd, int tok_major) {
  int bg = blockIdx.x, b = bg >> 5, g = bg & 31;
  int tid = threadIdx.x;
  float s = 0.f, s2 = 0.f;
  if (tok_major) {
    int c = g * 8 + (tid & 7);
    for (int hw = tid >> 3; hw < 1024; hw += 32) {
      float v = in[(((b << 10) + hw) << 8) + c];
      s += v; s2 += v * v;
    }
  } else {
    for (int i = tid; i < 8192; i += 256) {
      int c = g * 8 + (i >> 10), hw = i & 1023;
      float v = in[((b * 256 + c) << 10) + hw];
      s += v; s2 += v * v;
    }
  }
  __shared__ float sh[512];
  sh[tid] = s; sh[tid + 256] = s2;
  __syncthreads();
  for (int off = 128; off > 0; off >>= 1) {
    if (tid < off) { sh[tid] += sh[tid + off]; sh[tid + 256] += sh[tid + 256 + off]; }
    __syncthreads();
  }
  if (tid == 0) {
    float m = sh[0] * (1.f / 8192.f);
    float var = sh[256] * (1.f / 8192.f) - m * m;
    mean[bg] = m;
    rstd[bg] = rsqrtf(var + 1e-5f);
  }
}

// GN apply + swish, NCHW fp32 -> token-major bf16
__global__ void gn_apply_nchw_kernel(const float* __restrict__ in, const float* __restrict__ mean,
                                     const float* __restrict__ rstd, const float* __restrict__ sc,
                                     const float* __restrict__ bi, ushort_t* __restrict__ out) {
  int id = blockIdx.x * blockDim.x + threadIdx.x;
  if (id >= 16 * 256 * 1024) return;
  int hw = id & 1023, c = (id >> 10) & 255, b = id >> 18;
  int bg = b * 32 + (c >> 3);
  float v = (in[id] - mean[bg]) * rstd[bg] * sc[c] + bi[c];
  v = v / (1.f + expf(-v));
  out[(((b << 10) + hw) << 8) + c] = f32_to_bf16u(v);
}

// GN apply (+optional swish), token-major fp32 -> token-major bf16
__global__ void gn_apply_tok_kernel(const float* __restrict__ in, const float* __restrict__ mean,
                                    const float* __restrict__ rstd, const float* __restrict__ sc,
                                    const float* __restrict__ bi, ushort_t* __restrict__ out,
                                    int do_swish) {
  int id = blockIdx.x * blockDim.x + threadIdx.x;
  if (id >= 16 * 1024 * 256) return;
  int c = id & 255, t = id >> 8, b = t >> 10;
  int bg = b * 32 + (c >> 3);
  float v = (in[id] - mean[bg]) * rstd[bg] * sc[c] + bi[c];
  if (do_swish) v = v / (1.f + expf(-v));
  out[id] = f32_to_bf16u(v);
}

// ---------------------------------------------------------------------------
// Implicit-GEMM conv (3x3 or 1x1) with WMMA bf16.
// Block = 256 threads = 8 waves; each wave = one 16x16 output tile.
// MODE 0: fp32 tok out + bias + temb     MODE 1: fp32 tok out + bias + NCHW res
// MODE 2: qkv bf16 (qk[tok][512] / vT)   MODE 3: fp32 NCHW out + bias + tok res
// ---------------------------------------------------------------------------
template<int TAPS, int MODE>
__global__ __launch_bounds__(256)
void gemm_conv_kernel(const ushort_t* __restrict__ X, const ushort_t* __restrict__ Wp, int NT,
                      const float* __restrict__ bias, const float* __restrict__ aux,
                      float* __restrict__ outf, ushort_t* __restrict__ outq,
                      ushort_t* __restrict__ outv) {
  int lane = threadIdx.x & 31;
  int wv   = threadIdx.x >> 5;
  int m0   = (blockIdx.x * 2 + (wv & 1)) * 16;
  int ntg  = blockIdx.y * 4 + (wv >> 1);
  int n0   = ntg * 16;

  int tM  = m0 + (lane & 15);
  int b   = tM >> 10;
  int hw  = tM & 1023;
  int hh0 = hw >> 5, ww0 = hw & 31;

  v8f acc = zero8();
  #pragma unroll
  for (int tap = 0; tap < TAPS; ++tap) {
    int hh = hh0, ww = ww0;
    bool valid = true;
    if (TAPS == 9) {
      hh = hh0 + tap / 3 - 1;
      ww = ww0 + tap % 3 - 1;
      valid = ((unsigned)hh < 32u) && ((unsigned)ww < 32u);
    }
    const ushort_t* row = X + (size_t)(((b << 10) + (hh << 5) + ww) << 8);
    const ushort_t* wrow = Wp + ((size_t)((tap * 8) * NT + ntg) * 32 + lane) * 16;
    #pragma unroll
    for (int kt = 0; kt < 8; ++kt) {
      __builtin_prefetch(wrow + (size_t)NT * 512, 0, 1);   // next kt tile
      bf16x16 a  = valid ? load_a16(row, kt * 32, lane) : zero16();
      bf16x16 bw = load_wpack(wrow);
      acc = wmma_bf16(a, bw, acc);
      wrow += (size_t)NT * 512;
    }
  }

  // Epilogue: C/D layout -> VGPR r, lane l: M = r + (l<16?0:8), N = l&15
  int mhalf = (lane & 16) ? 8 : 0;
  int n = n0 + (lane & 15);
  #pragma unroll
  for (int r = 0; r < 8; ++r) {
    int tok = m0 + r + mhalf;
    int bb  = tok >> 10, thw = tok & 1023;
    float v = acc[r];
    if (MODE == 0) {
      v += bias[n] + aux[bb * 256 + n];
      outf[(size_t)tok * 256 + n] = v;
    } else if (MODE == 1) {
      v += bias[n] + aux[(size_t)((bb * 256 + n) << 10) + thw];
      outf[(size_t)tok * 256 + n] = v;
    } else if (MODE == 2) {
      if (n < 512) outq[(size_t)tok * 512 + n] = f32_to_bf16u(v);
      else         outv[((size_t)(bb * 256 + (n - 512)) << 10) + thw] = f32_to_bf16u(v);
    } else {
      v += bias[n] + aux[(size_t)tok * 256 + n];
      outf[((size_t)(bb * 256 + n) << 10) + thw] = v;
    }
  }
}

// ---------------------------------------------------------------------------
// Flash-style single-head attention. 256 threads = 8 waves per block.
// Wave = 16 queries x 128 output channels. K/V 32-key chunks are staged into
// LDS by the Tensor Data Mover (wave 0 issues D#s, double-buffered so the DMA
// of chunk kc+1 overlaps compute of chunk kc); all 8 waves feed their WMMAs
// from LDS. Online softmax uses half-wave shfl_xor row reductions; P is
// staged per-wave in LDS (same-wave DS ops are in-order) and re-read as an
// A fragment for the PV WMMAs.
// qk: [tok][512] bf16 (q at +0, k at +256). vT: [b*256+c][1024] bf16.
// ---------------------------------------------------------------------------
__global__ __launch_bounds__(256)
void attn_kernel(const ushort_t* __restrict__ qk, const ushort_t* __restrict__ vT,
                 ushort_t* __restrict__ outp) {
  __shared__ alignas(16) ushort_t kbuf[2][32 * 256];   // 2 x 16KB: 32 keys x 256ch
  __shared__ alignas(16) ushort_t vbuf[2][256 * 32];   // 2 x 16KB: 256ch x 32 keys
  __shared__ alignas(16) ushort_t pbuf[8][16 * 32];    // 8KB: per-wave P staging

  int lane  = threadIdx.x & 31;
  int wv    = threadIdx.x >> 5;
  int qt    = wv >> 1;
  int chalf = wv & 1;
  int m0    = blockIdx.x * 64 + qt * 16;
  int batch = m0 >> 10;
  int lh    = (lane & 16) ? 8 : 0;
  const float scale = 0.0625f;   // 1/sqrt(256)

  bf16x16 qf[8];
  {
    const ushort_t* qrow = qk + (size_t)(m0 + (lane & 15)) * 512;
    #pragma unroll
    for (int kt = 0; kt < 8; ++kt) qf[kt] = load_a16(qrow, kt * 32, lane);
  }

  v8f o[8];
  #pragma unroll
  for (int f = 0; f < 8; ++f) o[f] = zero8();
  float rm[8], rl[8];
  #pragma unroll
  for (int r = 0; r < 8; ++r) { rm[r] = -1e30f; rl[r] = 0.f; }

  ushort_t* pb = pbuf[wv];

#if HAVE_TDM
  unsigned koff[2] = { (unsigned)(size_t)&kbuf[0][0], (unsigned)(size_t)&kbuf[1][0] };
  unsigned voff[2] = { (unsigned)(size_t)&vbuf[0][0], (unsigned)(size_t)&vbuf[1][0] };
  // Prologue: DMA chunk 0 into buffer 0.
  if (wv == 0) {
    int j0 = batch * 1024;
    tdm_load_2d(koff[0], qk + (size_t)j0 * 512 + 256, 64, 32, 128);   // 32 x 512B, stride 1024B
    tdm_load_2d(voff[0], vT + ((size_t)(batch * 256) << 10), 8, 256, 256); // 256 x 64B, stride 2048B
  }
#endif

  int cur = 0;
  for (int kc = 0; kc < 32; ++kc) {
    __syncthreads();   // everyone done reading buffer cur^1 (overwritten next)
#if HAVE_TDM
    if (wv == 0) {
      if (kc + 1 < 32) {
        int j1 = batch * 1024 + (kc + 1) * 32;
        tdm_load_2d(koff[cur ^ 1], qk + (size_t)j1 * 512 + 256, 64, 32, 128);
        tdm_load_2d(voff[cur ^ 1], vT + ((size_t)(batch * 256) << 10) + (kc + 1) * 32,
                    8, 256, 256);
        __builtin_amdgcn_s_wait_tensorcnt(2);   // chunk kc landed (in-order)
      } else {
        __builtin_amdgcn_s_wait_tensorcnt(0);
      }
    }
#else
    // Fallback: cooperative staging of the K/V chunk with regular loads.
    {
      int j0 = batch * 1024 + kc * 32;
      for (int i = threadIdx.x; i < 1024; i += 256) {   // 1024 x uint4 = 16KB
        int row = i >> 5, c8 = i & 31;
        ((uint4*)kbuf[cur])[i] =
            *(const uint4*)(qk + (size_t)(j0 + row) * 512 + 256 + c8 * 8);
      }
      for (int i = threadIdx.x; i < 1024; i += 256) {
        int c = i >> 2, k8 = i & 3;
        ((uint4*)vbuf[cur])[i] =
            *(const uint4*)(vT + ((size_t)(batch * 256 + c) << 10) + kc * 32 + k8 * 8);
      }
    }
#endif
    __syncthreads();   // chunk kc visible to all waves

    const ushort_t* kb = kbuf[cur];
    const ushort_t* vb = vbuf[cur];

    // ---- S = q . k^T for 32 keys (two 16-key sub-tiles), B frags from LDS ----
    v8f s0 = zero8(), s1 = zero8();
    const ushort_t* krow0 = kb + (size_t)(lane & 15) * 256;
    const ushort_t* krow1 = kb + (size_t)(16 + (lane & 15)) * 256;
    #pragma unroll
    for (int kt = 0; kt < 8; ++kt) {
      s0 = wmma_bf16(qf[kt], load_b16(krow0, kt * 32, lane), s0);
      s1 = wmma_bf16(qf[kt], load_b16(krow1, kt * 32, lane), s1);
    }

    // ---- online softmax over this 32-key chunk ----
    float p0[8], p1[8], alpha[8];
    #pragma unroll
    for (int r = 0; r < 8; ++r) {
      float v0 = s0[r] * scale, v1 = s1[r] * scale;
      float mx = fmaxf(v0, v1);
      #pragma unroll
      for (int off = 1; off < 16; off <<= 1) mx = fmaxf(mx, __shfl_xor(mx, off, 32));
      float mnew = fmaxf(rm[r], mx);
      alpha[r] = expf(rm[r] - mnew);
      p0[r] = expf(v0 - mnew);
      p1[r] = expf(v1 - mnew);
      float rs = p0[r] + p1[r];
      #pragma unroll
      for (int off = 1; off < 16; off <<= 1) rs += __shfl_xor(rs, off, 32);
      rl[r] = rl[r] * alpha[r] + rs;
      rm[r] = mnew;
    }
    #pragma unroll
    for (int f = 0; f < 8; ++f)
      #pragma unroll
      for (int r = 0; r < 8; ++r) o[f][r] *= alpha[r];

    // ---- stage P (16x32 bf16) in this wave's LDS tile (in-order per-wave) ----
    #pragma unroll
    for (int r = 0; r < 8; ++r) {
      int M = r + lh;
      pb[M * 32 + (lane & 15)]      = f32_to_bf16u(p0[r]);
      pb[M * 32 + 16 + (lane & 15)] = f32_to_bf16u(p1[r]);
    }
    bf16x16 pa = load_a16(pb + (lane & 15) * 32, 0, lane);

    // ---- O += P . V  (8 channel tiles of 16), V frags from LDS ----
    #pragma unroll
    for (int f = 0; f < 8; ++f) {
      int c = chalf * 128 + f * 16 + (lane & 15);
      o[f] = wmma_bf16(pa, load_b16(vb + (size_t)c * 32, 0, lane), o[f]);
    }
    cur ^= 1;
  }

  // ---- normalize and store token-major bf16 ----
  #pragma unroll
  for (int f = 0; f < 8; ++f) {
    int c = chalf * 128 + f * 16 + (lane & 15);
    #pragma unroll
    for (int r = 0; r < 8; ++r) {
      int tok = m0 + r + lh;
      outp[(size_t)tok * 256 + c] = f32_to_bf16u(o[f][r] / rl[r]);
    }
  }
}

// ---------------------------------------------------------------------------
extern "C" void kernel_launch(void* const* d_in, const int* in_sizes, int n_in,
                              void* d_out, int out_size, void* d_ws, size_t ws_size,
                              hipStream_t stream) {
  (void)in_sizes; (void)n_in; (void)out_size; (void)ws_size;
  const float* x        = (const float*)d_in[0];
  const float* time_emb = (const float*)d_in[1];
  const float* gn1_s    = (const float*)d_in[2];
  const float* gn1_b    = (const float*)d_in[3];
  const float* conv1_w  = (const float*)d_in[4];
  const float* conv1_b  = (const float*)d_in[5];
  const float* noise_w  = (const float*)d_in[6];
  const float* noise_b  = (const float*)d_in[7];
  const float* gn2_s    = (const float*)d_in[8];
  const float* gn2_b    = (const float*)d_in[9];
  const float* conv2_w  = (const float*)d_in[10];
  const float* conv2_b  = (const float*)d_in[11];
  const float* agn_s    = (const float*)d_in[12];
  const float* agn_b    = (const float*)d_in[13];
  const float* qkv_w    = (const float*)d_in[14];
  const float* out_w    = (const float*)d_in[15];
  const float* out_b    = (const float*)d_in[16];
  float* out = (float*)d_out;

  char* p = (char*)d_ws;
  auto take = [&](size_t n) -> char* {
    char* r = p; p += (n + 255) & ~(size_t)255; return r;
  };
  float*    temb = (float*)take(16 * 256 * 4);
  float*    st_m = (float*)take(3 * 512 * 4);
  float*    st_r = (float*)take(3 * 512 * 4);
  ushort_t* wp1  = (ushort_t*)take((size_t)9 * 8 * 16 * 512 * 2);
  ushort_t* wp2  = (ushort_t*)take((size_t)9 * 8 * 16 * 512 * 2);
  ushort_t* wpq  = (ushort_t*)take((size_t)8 * 48 * 512 * 2);
  ushort_t* wpo  = (ushort_t*)take((size_t)8 * 16 * 512 * 2);
  ushort_t* xt   = (ushort_t*)take((size_t)16384 * 256 * 2);
  float*    h1   = (float*)take((size_t)16384 * 256 * 4);
  ushort_t* h1n  = (ushort_t*)take((size_t)16384 * 256 * 2);
  float*    hb   = (float*)take((size_t)16384 * 256 * 4);
  ushort_t* hn   = (ushort_t*)take((size_t)16384 * 256 * 2);
  ushort_t* qkb  = (ushort_t*)take((size_t)16384 * 512 * 2);
  ushort_t* vT   = (ushort_t*)take((size_t)16 * 256 * 1024 * 2);
  ushort_t* ao   = (ushort_t*)take((size_t)16384 * 256 * 2);

  const int TPB = 256;

  // Weight packing + time projection
  pack_w_kernel<<<(9 * 8 * 16 * 512 + TPB - 1) / TPB, TPB, 0, stream>>>(conv1_w, wp1, 16, 9);
  pack_w_kernel<<<(9 * 8 * 16 * 512 + TPB - 1) / TPB, TPB, 0, stream>>>(conv2_w, wp2, 16, 9);
  pack_w_kernel<<<(8 * 48 * 512 + TPB - 1) / TPB, TPB, 0, stream>>>(qkv_w, wpq, 48, 1);
  pack_w_kernel<<<(8 * 16 * 512 + TPB - 1) / TPB, TPB, 0, stream>>>(out_w, wpo, 16, 1);
  time_proj_kernel<<<16, 256, 0, stream>>>(time_emb, noise_w, noise_b, temb);

  // GN1 -> swish -> bf16 token-major
  gn_stats_kernel<<<512, 256, 0, stream>>>(x, st_m, st_r, 0);
  gn_apply_nchw_kernel<<<(16 * 256 * 1024) / TPB, TPB, 0, stream>>>(x, st_m, st_r, gn1_s, gn1_b, xt);

  // conv1 (+bias +temb) -> h1 fp32 token-major
  gemm_conv_kernel<9, 0><<<dim3(512, 4), TPB, 0, stream>>>(xt, wp1, 16, conv1_b, temb, h1, nullptr, nullptr);

  // GN2 -> swish -> bf16
  gn_stats_kernel<<<512, 256, 0, stream>>>(h1, st_m + 512, st_r + 512, 1);
  gn_apply_tok_kernel<<<(16 * 1024 * 256) / TPB, TPB, 0, stream>>>(h1, st_m + 512, st_r + 512, gn2_s, gn2_b, h1n, 1);

  // conv2 (+bias +x residual NCHW) -> h fp32 token-major
  gemm_conv_kernel<9, 1><<<dim3(512, 4), TPB, 0, stream>>>(h1n, wp2, 16, conv2_b, x, hb, nullptr, nullptr);

  // attention GN (no swish) -> bf16
  gn_stats_kernel<<<512, 256, 0, stream>>>(hb, st_m + 1024, st_r + 1024, 1);
  gn_apply_tok_kernel<<<(16 * 1024 * 256) / TPB, TPB, 0, stream>>>(hb, st_m + 1024, st_r + 1024, agn_s, agn_b, hn, 0);

  // qkv 1x1 conv -> qk [tok][512] bf16 + vT [b*256+c][1024] bf16
  gemm_conv_kernel<1, 2><<<dim3(512, 12), TPB, 0, stream>>>(hn, wpq, 48, nullptr, nullptr, nullptr, qkb, vT);

  // flash attention (TDM-staged K/V) -> ao bf16 token-major
  attn_kernel<<<256, 256, 0, stream>>>(qkb, vT, ao);

  // out 1x1 conv (+bias +h residual), store NCHW fp32 into d_out
  gemm_conv_kernel<1, 3><<<dim3(512, 4), TPB, 0, stream>>>(ao, wpo, 16, out_b, hb, out, nullptr, nullptr);
}
